// TMLPDGL_14027363189339
// MI455X (gfx1250) — matrix operand order repaired and checked by reading, not compiled, not run on hardware
//
#include <hip/hip_runtime.h>
#include <hip/hip_bf16.h>

typedef __attribute__((ext_vector_type(16))) __bf16 v16bf;
typedef __attribute__((ext_vector_type(8)))  __bf16 v8bf;
typedef __attribute__((ext_vector_type(8)))  float  v8f;

#define IN_DIM   64
#define HID      128
#define OUT_DIM  128
#define LN_EPS   1e-5f

// bf16 weight workspace layout (elements):
//   [0,8192)        W_e  [128][64]
//   [8192,16384)    W_s  [128][64]
//   [16384,24576)   W_d  [128][64]
//   [24576,40960)   W_o  [128][128]
#define WBF_ELEMS 40960
#define WBF_BYTES (WBF_ELEMS * 2)

__device__ __forceinline__ v16bf cvt16(float4 f0, float4 f1, float4 f2, float4 f3) {
    v16bf a;
    a[0]=(__bf16)f0.x;  a[1]=(__bf16)f0.y;  a[2]=(__bf16)f0.z;  a[3]=(__bf16)f0.w;
    a[4]=(__bf16)f1.x;  a[5]=(__bf16)f1.y;  a[6]=(__bf16)f1.z;  a[7]=(__bf16)f1.w;
    a[8]=(__bf16)f2.x;  a[9]=(__bf16)f2.y;  a[10]=(__bf16)f2.z; a[11]=(__bf16)f2.w;
    a[12]=(__bf16)f3.x; a[13]=(__bf16)f3.y; a[14]=(__bf16)f3.z; a[15]=(__bf16)f3.w;
    return a;
}

// 16 contiguous fp32 -> bf16 fragment (fallback path, inline convert)
__device__ __forceinline__ v16bf loadw16_f32(const float* __restrict__ p) {
    float4 f0 = *(const float4*)(p + 0);
    float4 f1 = *(const float4*)(p + 4);
    float4 f2 = *(const float4*)(p + 8);
    float4 f3 = *(const float4*)(p + 12);
    return cvt16(f0, f1, f2, f3);
}

// --------- prologue: one-shot fp32 -> bf16 weight conversion into ws -------
__global__ void __launch_bounds__(256)
convert_weights_kernel(const float* __restrict__ W_e,
                       const float* __restrict__ W_s,
                       const float* __restrict__ W_d,
                       const float* __restrict__ W_o,
                       __bf16* __restrict__ wbf)
{
    const int i = blockIdx.x * 256 + threadIdx.x;
    if (i >= WBF_ELEMS) return;
    float v;
    if      (i < 8192)  v = W_e[i];
    else if (i < 16384) v = W_s[i - 8192];
    else if (i < 24576) v = W_d[i - 16384];
    else                v = W_o[i - 24576];
    wbf[i] = (__bf16)v;
}

// --------- fused edge MLP: GEMM1(K=192,gather) + SiLU + GEMM2 + LayerNorm --
template <bool PREBF16>
__global__ void __launch_bounds__(128)
tmlp_edge_fused_kernel(const float* __restrict__ efeat,
                       const float* __restrict__ src_feat,
                       const float* __restrict__ dst_feat,
                       const int*   __restrict__ src_idx,
                       const int*   __restrict__ dst_idx,
                       const float* __restrict__ W_e,
                       const float* __restrict__ W_s,
                       const float* __restrict__ W_d,
                       const float* __restrict__ b1,
                       const float* __restrict__ W_o,
                       const float* __restrict__ b_out,
                       const float* __restrict__ gamma,
                       const float* __restrict__ beta,
                       const __bf16* __restrict__ wbf,
                       float* __restrict__ out)
{
    // per-wave LDS staging (4 waves / block)
    __shared__ __align__(16) __bf16 hb[4][16 * HID];      // silu(h) bf16 (A staging, GEMM2)
    __shared__ __align__(16) float  ho[4][16 * OUT_DIM];  // pre-LN output tile
    __shared__               float  st[4][16 * 2];        // per-row (mu, rsigma)

    const int lane = threadIdx.x & 31;
    const int wave = threadIdx.x >> 5;
    const int half = lane >> 4;      // 0: lanes 0-15, 1: lanes 16-31
    const int c    = lane & 15;      // A-frag row / B-frag column within 16-tile

    const int tile  = blockIdx.x * 4 + wave;   // 16 edges per wave
    const int ebase = tile * 16;
    const int e     = ebase + c;

    const int sidx = src_idx[e];
    const int didx = dst_idx[e];

    // ================= GEMM1: h = [efeat | src | dst] @ Wcat^T  (K = 192) ==
    v8f acc[8] = {};
    const int kbase = half * 8;

    #pragma unroll
    for (int kb = 0; kb < 6; ++kb) {
        const int sel = kb >> 1;             // 0: efeat, 1: src gather, 2: dst gather
        const int kk  = (kb & 1) * 32;       // K offset within the 64-wide source
        const float* abase =
            (sel == 0) ? (efeat    + (size_t)e    * IN_DIM) :
            (sel == 1) ? (src_feat + (size_t)sidx * IN_DIM) :
                         (dst_feat + (size_t)didx * IN_DIM);

        // A-frag bf16 layout: lane holds K = {k0..k0+15} of its row
        const int k0 = kk + kbase * 2;
        float4 f0 = *(const float4*)(abase + k0);
        float4 f1 = *(const float4*)(abase + k0 + 4);
        float4 f2 = *(const float4*)(abase + k0 + 8);
        float4 f3 = *(const float4*)(abase + k0 + 12);
        v16bf afrag = cvt16(f0, f1, f2, f3);

        const int kw = kk + half * 16;       // B-frag: 16 contiguous K of column n
        #pragma unroll
        for (int nb = 0; nb < 8; ++nb) {
            v16bf bfrag;
            if constexpr (PREBF16) {
                bfrag = *(const v16bf*)(wbf + (size_t)sel * 8192
                                        + (size_t)(nb * 16 + c) * IN_DIM + kw);
            } else {
                const float* Wsel = (sel == 0) ? W_e : (sel == 1) ? W_s : W_d;
                bfrag = loadw16_f32(Wsel + (size_t)(nb * 16 + c) * IN_DIM + kw);
            }
            acc[nb] = __builtin_amdgcn_wmma_f32_16x16x32_bf16(
                false, afrag, false, bfrag, (short)0, acc[nb], false, false);
        }
    }

    // ================= bias + SiLU, stage bf16 tile in LDS =================
    #pragma unroll
    for (int nb = 0; nb < 8; ++nb) {
        const float b1v = b1[nb * 16 + c];
        #pragma unroll
        for (int r = 0; r < 8; ++r) {
            const int m = r + 8 * half;                 // D-layout row
            float x = acc[nb][r] + b1v;
            x = x / (1.0f + __expf(-x));                // SiLU
            hb[wave][m * HID + nb * 16 + c] = (__bf16)x;
        }
    }
    __syncthreads();

    // ================= GEMM2: y = silu(h) @ W_out^T  (K = 128) =============
    v8f acc2[8] = {};
    #pragma unroll
    for (int kb = 0; kb < 4; ++kb) {
        const int k0 = kb * 32 + half * 16;
        v8bf lo = *(const v8bf*)&hb[wave][c * HID + k0];
        v8bf hi = *(const v8bf*)&hb[wave][c * HID + k0 + 8];
        v16bf afrag;
        #pragma unroll
        for (int i = 0; i < 8; ++i) { afrag[i] = lo[i]; afrag[8 + i] = hi[i]; }

        const int kw = kb * 32 + half * 16;
        #pragma unroll
        for (int nb = 0; nb < 8; ++nb) {
            v16bf bfrag;
            if constexpr (PREBF16) {
                bfrag = *(const v16bf*)(wbf + 24576
                                        + (size_t)(nb * 16 + c) * HID + kw);
            } else {
                bfrag = loadw16_f32(W_o + (size_t)(nb * 16 + c) * HID + kw);
            }
            acc2[nb] = __builtin_amdgcn_wmma_f32_16x16x32_bf16(
                false, afrag, false, bfrag, (short)0, acc2[nb], false, false);
        }
    }

    // ================= + b_out, row stats via shfl_xor, stage f32 tile =====
    float bo[8];
    #pragma unroll
    for (int nb = 0; nb < 8; ++nb) bo[nb] = b_out[nb * 16 + c];

    #pragma unroll
    for (int r = 0; r < 8; ++r) {
        const int m = r + 8 * half;
        float ps = 0.f, pq = 0.f;
        #pragma unroll
        for (int nb = 0; nb < 8; ++nb) {
            float x = acc2[nb][r] + bo[nb];
            ho[wave][m * OUT_DIM + nb * 16 + c] = x;
            ps += x; pq += x * x;
        }
        // each 16-lane half covers one full 128-wide row -> xor-reduce 1,2,4,8
        #pragma unroll
        for (int off = 1; off <= 8; off <<= 1) {
            ps += __shfl_xor(ps, off);
            pq += __shfl_xor(pq, off);
        }
        if (c == 0) {
            const float mu  = ps * (1.0f / OUT_DIM);
            const float var = pq * (1.0f / OUT_DIM) - mu * mu;
            st[wave][m * 2 + 0] = mu;
            st[wave][m * 2 + 1] = rsqrtf(var + LN_EPS);
        }
    }
    __syncthreads();

    // ================= LayerNorm epilogue: coalesced float4 stores =========
    const int orow = lane >> 1;          // 2 lanes per row, 64 cols each
    const int cb   = (lane & 1) * 64;
    const float mu = st[wave][orow * 2 + 0];
    const float rs = st[wave][orow * 2 + 1];
    float* op = out + (size_t)(ebase + orow) * OUT_DIM;
    #pragma unroll
    for (int j = 0; j < 16; ++j) {
        const int col = cb + j * 4;
        float4 x = *(const float4*)&ho[wave][orow * OUT_DIM + col];
        float4 g = *(const float4*)&gamma[col];
        float4 b = *(const float4*)&beta[col];
        float4 y;
        y.x = (x.x - mu) * rs * g.x + b.x;
        y.y = (x.y - mu) * rs * g.y + b.y;
        y.z = (x.z - mu) * rs * g.z + b.z;
        y.w = (x.w - mu) * rs * g.w + b.w;
        *(float4*)(op + col) = y;
    }
}

extern "C" void kernel_launch(void* const* d_in, const int* in_sizes, int n_in,
                              void* d_out, int out_size, void* d_ws, size_t ws_size,
                              hipStream_t stream)
{
    const float* efeat    = (const float*)d_in[0];
    const float* src_feat = (const float*)d_in[1];
    const float* dst_feat = (const float*)d_in[2];
    const int*   src_idx  = (const int*)d_in[3];
    const int*   dst_idx  = (const int*)d_in[4];
    const float* W_e      = (const float*)d_in[5];
    const float* W_s      = (const float*)d_in[6];
    const float* W_d      = (const float*)d_in[7];
    const float* b1       = (const float*)d_in[8];
    const float* W_o      = (const float*)d_in[9];
    const float* b_out    = (const float*)d_in[10];
    const float* gamma    = (const float*)d_in[11];
    const float* beta     = (const float*)d_in[12];
    float* out = (float*)d_out;

    const int n_edges = in_sizes[0] / IN_DIM;   // 1,000,000
    const int blocks  = n_edges / 64;           // 64 edges per block (4 waves x 16)

    if (ws_size >= (size_t)WBF_BYTES) {
        __bf16* wbf = (__bf16*)d_ws;
        convert_weights_kernel<<<(WBF_ELEMS + 255) / 256, 256, 0, stream>>>(
            W_e, W_s, W_d, W_o, wbf);
        tmlp_edge_fused_kernel<true><<<blocks, 128, 0, stream>>>(
            efeat, src_feat, dst_feat, src_idx, dst_idx,
            W_e, W_s, W_d, b1, W_o, b_out, gamma, beta, wbf, out);
    } else {
        tmlp_edge_fused_kernel<false><<<blocks, 128, 0, stream>>>(
            efeat, src_feat, dst_feat, src_idx, dst_idx,
            W_e, W_s, W_d, b1, W_o, b_out, gamma, beta, nullptr, out);
    }
}
